// AttentionBlock_59682865545843
// MI455X (gfx1250) — matrix operand order repaired and checked by reading, not compiled, not run on hardware
//
#include <hip/hip_runtime.h>

// ---------------------------------------------------------------------------
// AttentionBlock for MI455X (gfx1250): bf16 WMMA + async LDS staging.
//   K1: fp32 -> bf16 convert (+ transpose W_in, W_out)
//   K2: proj = X @ W_in  (WMMA, 32x64 per wave, async B double buffer) + RoPE
//   K3: flash attention: async K/V tile staging, QK^T WMMA, online softmax,
//       P via LDS, P@V WMMA
//   K4: out = O @ W_out  (WMMA, 32x64 per wave, async B double buffer), fp32
// ---------------------------------------------------------------------------

#define BATCH 2
#define SEQ   2048
#define DM    1024
#define NH    16
#define HD    64
#define NTOK  (BATCH * SEQ)   // 4096
#define W3    (3 * DM)        // 3072

typedef __attribute__((ext_vector_type(16))) __bf16 v16bf;
typedef __attribute__((ext_vector_type(8)))  __bf16 v8bf;
typedef __attribute__((ext_vector_type(8)))  float  v8f;

union AFrag { v16bf v; v8bf h[2]; };

// ln(10000)/32  (RoPE half = 32)
#define ROPE_LN_SCALE 0.2878231366242991f

// Async copy 16B/lane from global to LDS (CDNA5, tracked by ASYNCcnt).
// Flat LDS aperture keeps the LDS offset in addr[31:0] (ISA 10.2), so the
// truncated generic pointer is a valid per-lane LDS address VGPR.
__device__ __forceinline__ void async_b128(const void* g, void* l) {
  unsigned lds = (unsigned)(size_t)l;
  asm volatile("global_load_async_to_lds_b128 %0, %1, off"
               :: "v"(lds), "v"(g) : "memory");
}
#define WAIT_ASYNC_0() asm volatile("s_wait_asynccnt 0x0" ::: "memory")
#define WAIT_ASYNC_2() asm volatile("s_wait_asynccnt 0x2" ::: "memory")

#define WMMA_BF16(A, B, C) \
  __builtin_amdgcn_wmma_f32_16x16x32_bf16(false, (A), false, (B), (short)0, (C), false, false)

// ---------------------------------------------------------------------------
// K1: convert inputs to bf16; transpose weights so GEMM B-operands are
// contraction-contiguous per output column (perfect WMMA B-fragment loads).
// ---------------------------------------------------------------------------
__global__ __launch_bounds__(256) void k_convert(
    const float* __restrict__ X, const float* __restrict__ Win,
    const float* __restrict__ Wout,
    __bf16* __restrict__ Xb, __bf16* __restrict__ WinT, __bf16* __restrict__ WoutT)
{
  int i = blockIdx.x * blockDim.x + threadIdx.x;
  int stride = gridDim.x * blockDim.x;
  for (int idx = i; idx < NTOK * DM; idx += stride)
    Xb[idx] = (__bf16)X[idx];
  for (int idx = i; idx < DM * W3; idx += stride) {
    int k = idx / W3, n = idx % W3;
    WinT[n * DM + k] = (__bf16)Win[idx];
  }
  for (int idx = i; idx < DM * DM; idx += stride) {
    int k = idx / DM, n = idx % DM;
    WoutT[n * DM + k] = (__bf16)Wout[idx];
  }
}

// ---------------------------------------------------------------------------
// K2: projection GEMM + fused RoPE. grid = (NTOK/256, 48).
// Wave tile: 32 rows x 64 cols, K-step 64 -> 16 WMMAs per sync step.
// B tile (64 cols x 64 K, 8KB) staged in LDS via two async copies,
// double-buffered, shared by all 8 waves.
// ---------------------------------------------------------------------------
__global__ __launch_bounds__(256) void k_proj_rope(
    const __bf16* __restrict__ Xb, const __bf16* __restrict__ WinT,
    const int* __restrict__ pos,
    __bf16* __restrict__ Qb, __bf16* __restrict__ Kb, __bf16* __restrict__ Vt)
{
  __shared__ __bf16 bt[2][64][72];   // padded stride: 144B (16B-aligned rows)

  const int tid = threadIdx.x;
  const int wave = tid >> 5, lane = tid & 31;
  const int hl = lane >> 4, ln = lane & 15;
  const int m0 = blockIdx.x * 256 + wave * 32;
  const int n0 = blockIdx.y * 64;
  const int head = blockIdx.y / 3, type = blockIdx.y % 3;

  // async-copy mapping: 512 pieces of 16B over 64 rows x 128B; 2 issues/step
  const int brow = tid >> 3, bc = (tid & 7) * 8;
  const __bf16* bsrc0 = WinT + (size_t)(n0 + brow) * DM + bc;
  const __bf16* bsrc1 = WinT + (size_t)(n0 + 32 + brow) * DM + bc;

  async_b128(bsrc0, &bt[0][brow][bc]);
  async_b128(bsrc1, &bt[0][32 + brow][bc]);

  v8f acc[2][4] = {};
  const __bf16* arow0 = Xb + (size_t)(m0 + ln) * DM;
  const __bf16* arow1 = Xb + (size_t)(m0 + 16 + ln) * DM;

  const int NSTEP = DM / 64;  // 16
  for (int i = 0; i < NSTEP; ++i) {
    const int cur = i & 1;
    __syncthreads();  // everyone finished reading buffer (1-cur)
    if (i + 1 < NSTEP) {
      async_b128(bsrc0 + (size_t)(i + 1) * 64, &bt[1 - cur][brow][bc]);
      async_b128(bsrc1 + (size_t)(i + 1) * 64, &bt[1 - cur][32 + brow][bc]);
      WAIT_ASYNC_2();  // current tile landed; next two stay in flight
    } else {
      WAIT_ASYNC_0();
    }
    __syncthreads();  // current tile visible to all waves

    const int k0 = i * 64;
#pragma unroll
    for (int kk = 0; kk < 2; ++kk) {
      const int ko = k0 + kk * 32;
      AFrag a0, a1;
      a0.h[0] = *(const v8bf*)(arow0 + ko + 8 * hl);
      a0.h[1] = *(const v8bf*)(arow0 + ko + 16 + 8 * hl);
      a1.h[0] = *(const v8bf*)(arow1 + ko + 8 * hl);
      a1.h[1] = *(const v8bf*)(arow1 + ko + 16 + 8 * hl);
#pragma unroll
      for (int t = 0; t < 4; ++t) {
        AFrag b;
        b.h[0] = *(const v8bf*)(&bt[cur][16 * t + ln][32 * kk + 8 * hl]);
        b.h[1] = *(const v8bf*)(&bt[cur][16 * t + ln][32 * kk + 16 + 8 * hl]);
        acc[0][t] = WMMA_BF16(a0.v, b.v, acc[0][t]);
        acc[1][t] = WMMA_BF16(a1.v, b.v, acc[1][t]);
      }
    }
  }

  if (type < 2) {  // RoPE on q and k: pairs (j, j+32) are tiles (t, t+2)
#pragma unroll
    for (int rb = 0; rb < 2; ++rb) {
#pragma unroll
      for (int t = 0; t < 2; ++t) {
        float j = (float)(16 * t + ln);
        float inv_freq = __expf(-j * ROPE_LN_SCALE);
#pragma unroll
        for (int d = 0; d < 8; ++d) {
          int row = m0 + 16 * rb + d + 8 * hl;
          float ang = (float)pos[row] * inv_freq;
          float c = __cosf(ang), s = __sinf(ang);
          float x1 = acc[rb][t][d], x2 = acc[rb][t + 2][d];
          acc[rb][t][d]     = x1 * c - x2 * s;
          acc[rb][t + 2][d] = x1 * s + x2 * c;
        }
      }
    }
  }

#pragma unroll
  for (int rb = 0; rb < 2; ++rb) {
#pragma unroll
    for (int t = 0; t < 4; ++t) {
#pragma unroll
      for (int d = 0; d < 8; ++d) {
        int row = m0 + 16 * rb + d + 8 * hl;
        int b = row >> 11, s = row & (SEQ - 1);
        int dim = 16 * t + ln;
        __bf16 val = (__bf16)acc[rb][t][d];
        size_t bh = (size_t)(b * NH + head);
        if (type == 2)      Vt[(bh * HD + dim) * SEQ + s] = val;  // V transposed
        else if (type == 0) Qb[(bh * SEQ + s) * HD + dim] = val;
        else                Kb[(bh * SEQ + s) * HD + dim] = val;
      }
    }
  }
}

// ---------------------------------------------------------------------------
// K3: flash attention. grid = (SEQ/128, BATCH*NH). Wave = 16 query rows.
// K tile (32 keys x 64d) and V^T tile (64d x 32 keys) staged via async copy,
// double-buffered, shared by all 8 waves. Per 32-key step: 4 WMMA (QK^T),
// shfl_xor online softmax, P through per-wave LDS, 4 WMMA (P@V).
// ---------------------------------------------------------------------------
__global__ __launch_bounds__(256) void k_attn(
    const __bf16* __restrict__ Qb, const __bf16* __restrict__ Kb,
    const __bf16* __restrict__ Vt, __bf16* __restrict__ Ob)
{
  __shared__ __bf16 kt[2][32][72];   // 32 keys x 64 dims, stride 144B
  __shared__ __bf16 vt[2][64][40];   // 64 dims  x 32 keys, stride 80B
  __shared__ __bf16 pbuf[8][16][48]; // per-wave P tile, stride 96B

  const int tid = threadIdx.x;
  const int wave = tid >> 5, lane = tid & 31;
  const int hl = lane >> 4, ln = lane & 15;
  const int bh = blockIdx.y;
  const int q0 = blockIdx.x * 128 + wave * 16;

  const __bf16* Qh = Qb + (size_t)bh * SEQ * HD;
  const __bf16* Kh = Kb + (size_t)bh * SEQ * HD;
  const __bf16* Vh = Vt + (size_t)bh * HD * SEQ;

  // async-copy mappings (256 threads x 16B per tile)
  const int krow = tid >> 3, kc = (tid & 7) * 8;  // 32 x 128B
  const int vrow = tid >> 2, vc = (tid & 3) * 8;  // 64 x 64B
  const __bf16* ksrc = Kh + (size_t)krow * HD + kc;
  const __bf16* vsrc = Vh + (size_t)vrow * SEQ + vc;

  async_b128(ksrc, &kt[0][krow][kc]);
  async_b128(vsrc, &vt[0][vrow][vc]);

  AFrag aq0, aq1;  // Q rows, head-dim contraction 0..31 and 32..63
  const __bf16* qrow = Qh + (size_t)(q0 + ln) * HD;
  aq0.h[0] = *(const v8bf*)(qrow + 8 * hl);
  aq0.h[1] = *(const v8bf*)(qrow + 16 + 8 * hl);
  aq1.h[0] = *(const v8bf*)(qrow + 32 + 8 * hl);
  aq1.h[1] = *(const v8bf*)(qrow + 48 + 8 * hl);

  v8f o[4] = {};
  float mrow[8], lrow[8];
#pragma unroll
  for (int d = 0; d < 8; ++d) { mrow[d] = -1e30f; lrow[d] = 0.0f; }

  for (int i = 0; i < SEQ / 32; ++i) {
    const int cur = i & 1;
    __syncthreads();  // everyone finished reading buffer (1-cur)
    if (i + 1 < SEQ / 32) {
      async_b128(ksrc + (size_t)(i + 1) * 32 * HD, &kt[1 - cur][krow][kc]);
      async_b128(vsrc + (size_t)(i + 1) * 32,      &vt[1 - cur][vrow][vc]);
      WAIT_ASYNC_2();  // current K/V landed; next pair stays in flight
    } else {
      WAIT_ASYNC_0();
    }
    __syncthreads();

    // ---- scores: two 16x16 key tiles, contraction = head dim (2 chunks) ----
    v8f s0 = {}, s1 = {};
    {
      AFrag b00, b01, b10, b11;
      b00.h[0] = *(const v8bf*)(&kt[cur][ln][8 * hl]);
      b00.h[1] = *(const v8bf*)(&kt[cur][ln][16 + 8 * hl]);
      b01.h[0] = *(const v8bf*)(&kt[cur][ln][32 + 8 * hl]);
      b01.h[1] = *(const v8bf*)(&kt[cur][ln][48 + 8 * hl]);
      b10.h[0] = *(const v8bf*)(&kt[cur][16 + ln][8 * hl]);
      b10.h[1] = *(const v8bf*)(&kt[cur][16 + ln][16 + 8 * hl]);
      b11.h[0] = *(const v8bf*)(&kt[cur][16 + ln][32 + 8 * hl]);
      b11.h[1] = *(const v8bf*)(&kt[cur][16 + ln][48 + 8 * hl]);
      s0 = WMMA_BF16(aq0.v, b00.v, s0);
      s0 = WMMA_BF16(aq1.v, b01.v, s0);
      s1 = WMMA_BF16(aq0.v, b10.v, s1);
      s1 = WMMA_BF16(aq1.v, b11.v, s1);
    }

    // ---- online softmax (rows live at VGPR d, lane-half hl) ----
    float pm0[8], pm1[8];
#pragma unroll
    for (int d = 0; d < 8; ++d) {
      float a0 = s0[d] * 0.125f, a1 = s1[d] * 0.125f;  // 1/sqrt(64)
      float mx = fmaxf(a0, a1);
#pragma unroll
      for (int m = 1; m < 16; m <<= 1) mx = fmaxf(mx, __shfl_xor(mx, m, 32));
      float mnew = fmaxf(mrow[d], mx);
      float alpha = __expf(mrow[d] - mnew);
      float p0 = __expf(a0 - mnew), p1 = __expf(a1 - mnew);
      float sum = p0 + p1;
#pragma unroll
      for (int m = 1; m < 16; m <<= 1) sum += __shfl_xor(sum, m, 32);
      lrow[d] = lrow[d] * alpha + sum;
      mrow[d] = mnew;
      pm0[d] = p0; pm1[d] = p1;
#pragma unroll
      for (int tn = 0; tn < 4; ++tn) o[tn][d] *= alpha;
    }

    // ---- stage P (C/D layout) -> per-wave LDS -> reload as A fragment ----
    // (same-wave DS ops are in-order on CDNA5: no barrier needed)
#pragma unroll
    for (int d = 0; d < 8; ++d) {
      pbuf[wave][d + 8 * hl][ln]      = (__bf16)pm0[d];
      pbuf[wave][d + 8 * hl][16 + ln] = (__bf16)pm1[d];
    }
    AFrag ap;  // 16x32: rows = q, contraction = 32 keys of this step
    ap.h[0] = *(const v8bf*)(&pbuf[wave][ln][8 * hl]);
    ap.h[1] = *(const v8bf*)(&pbuf[wave][ln][16 + 8 * hl]);

    // ---- O += P @ V ----
#pragma unroll
    for (int tn = 0; tn < 4; ++tn) {
      AFrag bv;
      bv.h[0] = *(const v8bf*)(&vt[cur][16 * tn + ln][8 * hl]);
      bv.h[1] = *(const v8bf*)(&vt[cur][16 * tn + ln][16 + 8 * hl]);
      o[tn] = WMMA_BF16(ap.v, bv.v, o[tn]);
    }
  }

  // ---- normalize and scatter to O buffer [NTOK, DM] for the final GEMM ----
  const int hcol = (bh & (NH - 1)) * HD;
  const int rowbase = (bh >> 4) * SEQ + q0;
#pragma unroll
  for (int tn = 0; tn < 4; ++tn) {
#pragma unroll
    for (int d = 0; d < 8; ++d) {
      float v = o[tn][d] / lrow[d];
      Ob[(size_t)(rowbase + d + 8 * hl) * DM + hcol + tn * 16 + ln] = (__bf16)v;
    }
  }
}

// ---------------------------------------------------------------------------
// K4: out = O @ W_out, fp32 result. grid = (NTOK/256, DM/64).
// Wave tile 32x64, K-step 64, async double-buffered B staging (as K2).
// ---------------------------------------------------------------------------
__global__ __launch_bounds__(256) void k_out_gemm(
    const __bf16* __restrict__ Ob, const __bf16* __restrict__ WoutT,
    float* __restrict__ out)
{
  __shared__ __bf16 bt[2][64][72];

  const int tid = threadIdx.x;
  const int wave = tid >> 5, lane = tid & 31;
  const int hl = lane >> 4, ln = lane & 15;
  const int m0 = blockIdx.x * 256 + wave * 32;
  const int n0 = blockIdx.y * 64;

  const int brow = tid >> 3, bc = (tid & 7) * 8;
  const __bf16* bsrc0 = WoutT + (size_t)(n0 + brow) * DM + bc;
  const __bf16* bsrc1 = WoutT + (size_t)(n0 + 32 + brow) * DM + bc;

  async_b128(bsrc0, &bt[0][brow][bc]);
  async_b128(bsrc1, &bt[0][32 + brow][bc]);

  v8f acc[2][4] = {};
  const __bf16* arow0 = Ob + (size_t)(m0 + ln) * DM;
  const __bf16* arow1 = Ob + (size_t)(m0 + 16 + ln) * DM;

  const int NSTEP = DM / 64;
  for (int i = 0; i < NSTEP; ++i) {
    const int cur = i & 1;
    __syncthreads();
    if (i + 1 < NSTEP) {
      async_b128(bsrc0 + (size_t)(i + 1) * 64, &bt[1 - cur][brow][bc]);
      async_b128(bsrc1 + (size_t)(i + 1) * 64, &bt[1 - cur][32 + brow][bc]);
      WAIT_ASYNC_2();
    } else {
      WAIT_ASYNC_0();
    }
    __syncthreads();

    const int k0 = i * 64;
#pragma unroll
    for (int kk = 0; kk < 2; ++kk) {
      const int ko = k0 + kk * 32;
      AFrag a0, a1;
      a0.h[0] = *(const v8bf*)(arow0 + ko + 8 * hl);
      a0.h[1] = *(const v8bf*)(arow0 + ko + 16 + 8 * hl);
      a1.h[0] = *(const v8bf*)(arow1 + ko + 8 * hl);
      a1.h[1] = *(const v8bf*)(arow1 + ko + 16 + 8 * hl);
#pragma unroll
      for (int t = 0; t < 4; ++t) {
        AFrag b;
        b.h[0] = *(const v8bf*)(&bt[cur][16 * t + ln][32 * kk + 8 * hl]);
        b.h[1] = *(const v8bf*)(&bt[cur][16 * t + ln][32 * kk + 16 + 8 * hl]);
        acc[0][t] = WMMA_BF16(a0.v, b.v, acc[0][t]);
        acc[1][t] = WMMA_BF16(a1.v, b.v, acc[1][t]);
      }
    }
  }
#pragma unroll
  for (int rb = 0; rb < 2; ++rb)
#pragma unroll
    for (int t = 0; t < 4; ++t)
#pragma unroll
      for (int d = 0; d < 8; ++d)
        out[(size_t)(m0 + 16 * rb + d + 8 * hl) * DM + n0 + 16 * t + ln] = acc[rb][t][d];
}

// ---------------------------------------------------------------------------
extern "C" void kernel_launch(void* const* d_in, const int* in_sizes, int n_in,
                              void* d_out, int out_size, void* d_ws, size_t ws_size,
                              hipStream_t stream) {
  const float* X    = (const float*)d_in[0];
  const int*   pos  = (const int*)d_in[1];
  const float* Win  = (const float*)d_in[2];
  const float* Wout = (const float*)d_in[3];
  float* out = (float*)d_out;

  __bf16* Xb    = (__bf16*)d_ws;
  __bf16* WinT  = Xb    + (size_t)NTOK * DM;
  __bf16* WoutT = WinT  + (size_t)W3 * DM;
  __bf16* Qb    = WoutT + (size_t)DM * DM;
  __bf16* Kb    = Qb    + (size_t)BATCH * NH * SEQ * HD;
  __bf16* Vt    = Kb    + (size_t)BATCH * NH * SEQ * HD;
  __bf16* Ob    = Vt    + (size_t)BATCH * NH * SEQ * HD;
  // total: 48 MB bf16 workspace

  k_convert<<<2048, 256, 0, stream>>>(X, Win, Wout, Xb, WinT, WoutT);
  k_proj_rope<<<dim3(NTOK / 256, W3 / 64), 256, 0, stream>>>(Xb, WinT, pos, Qb, Kb, Vt);
  k_attn<<<dim3(SEQ / 128, BATCH * NH), 256, 0, stream>>>(Qb, Kb, Vt, Ob);
  k_out_gemm<<<dim3(NTOK / 256, DM / 64), 256, 0, stream>>>(Ob, WoutT, out);
}
